// MultiHeadedSelfAttention_16638703305310
// MI455X (gfx1250) — matrix operand order repaired and checked by reading, not compiled, not run on hardware
//
#include <hip/hip_runtime.h>

typedef __attribute__((ext_vector_type(16))) __bf16 v16bf;
typedef __attribute__((ext_vector_type(8)))  float  v8f;

#define DM   1024
#define SEQ  2048
#define NB   2
#define NH   16
#define DH   64
#define MTOT (NB*SEQ)   // 4096

// ---------- WMMA helpers ----------
__device__ __forceinline__ v8f wmma_bf16(v16bf a, v16bf b, v8f c) {
  return __builtin_amdgcn_wmma_f32_16x16x32_bf16(false, a, false, b, (short)0, c,
                                                 false, false);
}

// A-operand (16x32 bf16): lane holds row M=lane%16.
// per-lane K order: j<8 -> K=half*8+j ; j>=8 -> K=16+half*8+(j-8)
__device__ __forceinline__ v16bf load_fragA(const __bf16* p, int half) {
  union { v16bf v; uint4 u[2]; } f;
  f.u[0] = *reinterpret_cast<const uint4*>(p + half * 8);
  f.u[1] = *reinterpret_cast<const uint4*>(p + 16 + half * 8);
  return f.v;
}

// B-operand (32x16 bf16): lane holds column N=lane%16 (a row of B^T).
// per-lane K order: j<8 -> K=half*8+j ; j>=8 -> K=16+half*8+(j-8)
__device__ __forceinline__ v16bf load_fragB(const __bf16* p, int half) {
  union { v16bf v; uint4 u[2]; } f;
  f.u[0] = *reinterpret_cast<const uint4*>(p + half * 8);
  f.u[1] = *reinterpret_cast<const uint4*>(p + 16 + half * 8);
  return f.v;
}

// ---------- CDNA5 async global->LDS (ASYNCcnt-tracked, no VGPR round trip) ----
// LDS offset = low 32 bits of the generic pointer (aperture keeps offset in [31:0])
__device__ __forceinline__ void async_ld128(void* lds_ptr, const void* g) {
  unsigned lds_off = (unsigned)(uintptr_t)lds_ptr;
  asm volatile("global_load_async_to_lds_b128 %0, %1, off"
               :: "v"(lds_off), "v"(g) : "memory");
}
__device__ __forceinline__ void wait_async0() {
  asm volatile("s_wait_asynccnt 0x0" ::: "memory");
}

// ---------- conversion kernels ----------
__global__ void cvt_kernel(const float* __restrict__ in, __bf16* __restrict__ out, int n) {
  int i = blockIdx.x * 256 + threadIdx.x;
  if (i < n) out[i] = (__bf16)in[i];
}

// w[k][n] (1024x1024 f32) -> wT[n][k] (bf16)
__global__ void cvtT_kernel(const float* __restrict__ w, __bf16* __restrict__ wT) {
  int i = blockIdx.x * 256 + threadIdx.x;
  int k = i >> 10, n = i & 1023;
  wT[(size_t)n * DM + k] = (__bf16)w[i];
}

// ---------- GEMM: C[M][1024] = A[M][1024] * BT^T + bias ----------
// mode 0: bf16 row-major out; mode 1: f32 row-major out;
// mode 2: bf16 out transposed per head: [b][h][dh][SEQ]  (for V)
__launch_bounds__(256)
__global__ void gemm_bf16_kernel(const __bf16* __restrict__ A,
                                 const __bf16* __restrict__ BT,
                                 const float* __restrict__ bias,
                                 void* __restrict__ out, int mode) {
  __shared__ __bf16 sA[2][128 * 40];   // 128 rows x 32 K, stride 40 (conflict-free)
  __shared__ __bf16 sB[2][128 * 40];
  const int tid = threadIdx.x;
  const int lane = tid & 31, w = tid >> 5;
  const int ln = lane & 15, lh = lane >> 4;
  const int m0 = blockIdx.y * 128, n0 = blockIdx.x * 128;
  const int wr = w >> 1, wc = w & 1;    // wave: rows [wr*32,+32), cols [wc*64,+64)

  const int lrow = tid >> 2, lch = tid & 3;  // staging: 128 rows x 4 chunks, x2
  auto stage = [&](int kt, int buf) {
    const __bf16* ga = A  + (size_t)(m0 + lrow) * DM + kt * 32 + lch * 8;
    const __bf16* gb = BT + (size_t)(n0 + lrow) * DM + kt * 32 + lch * 8;
    async_ld128(&sA[buf][lrow * 40 + lch * 8], ga);
    async_ld128(&sA[buf][(lrow + 64) * 40 + lch * 8], ga + (size_t)64 * DM);
    async_ld128(&sB[buf][lrow * 40 + lch * 8], gb);
    async_ld128(&sB[buf][(lrow + 64) * 40 + lch * 8], gb + (size_t)64 * DM);
  };

  const v8f vz = {0.f, 0.f, 0.f, 0.f, 0.f, 0.f, 0.f, 0.f};
  v8f acc[2][4];
#pragma unroll
  for (int mi = 0; mi < 2; ++mi)
#pragma unroll
    for (int ni = 0; ni < 4; ++ni) acc[mi][ni] = vz;

  stage(0, 0);
  for (int kt = 0; kt < DM / 32; ++kt) {
    const int buf = kt & 1;
    wait_async0();
    __syncthreads();
    if (kt + 1 < DM / 32) stage(kt + 1, buf ^ 1);

    v16bf af[2], bfm[4];
#pragma unroll
    for (int mi = 0; mi < 2; ++mi)
      af[mi] = load_fragA(&sA[buf][(wr * 32 + mi * 16 + ln) * 40], lh);
#pragma unroll
    for (int ni = 0; ni < 4; ++ni)
      bfm[ni] = load_fragB(&sB[buf][(wc * 64 + ni * 16 + ln) * 40], lh);
#pragma unroll
    for (int mi = 0; mi < 2; ++mi)
#pragma unroll
      for (int ni = 0; ni < 4; ++ni)
        acc[mi][ni] = wmma_bf16(af[mi], bfm[ni], acc[mi][ni]);
  }

#pragma unroll
  for (int mi = 0; mi < 2; ++mi) {
#pragma unroll
    for (int ni = 0; ni < 4; ++ni) {
      const int gn = n0 + wc * 64 + ni * 16 + ln;
      const float bv = bias[gn];
      const int gm = m0 + wr * 32 + mi * 16 + lh * 8;  // 8 consecutive rows
      if (mode == 0) {
        __bf16* o = (__bf16*)out;
#pragma unroll
        for (int r = 0; r < 8; ++r)
          o[(size_t)(gm + r) * DM + gn] = (__bf16)(acc[mi][ni][r] + bv);
      } else if (mode == 1) {
        float* o = (float*)out;
#pragma unroll
        for (int r = 0; r < 8; ++r)
          o[(size_t)(gm + r) * DM + gn] = acc[mi][ni][r] + bv;
      } else {
        __bf16* o = (__bf16*)out;
        const int h = gn >> 6, d = gn & 63;
        const int bb = gm >> 11, s = gm & 2047;  // 128-row tiles never cross batch
        __bf16* orow = o + ((size_t)(bb * NH + h) * DH + d) * SEQ + s;
#pragma unroll
        for (int r = 0; r < 8; ++r) orow[r] = (__bf16)(acc[mi][ni][r] + bv);
      }
    }
  }
}

// ---------- flash attention ----------
// Q,K: bf16 [b][s][1024] (head-contiguous dh=64); Vt: bf16 [b][h][64][2048]
// O: bf16 [b][s][1024]
__launch_bounds__(256)
__global__ void attn_kernel(const __bf16* __restrict__ Q, const __bf16* __restrict__ Kg,
                            const __bf16* __restrict__ Vt, __bf16* __restrict__ O) {
  __shared__ __bf16 sK[2][128 * 72];      // 128 keys x 64 dh
  __shared__ __bf16 sV[2][64 * 136];      // 64 dh   x 128 keys
  __shared__ __bf16 sP[8 * 16 * 136];     // per wave: 16 q x 128 keys
  const int tid = threadIdx.x;
  const int lane = tid & 31, w = tid >> 5;
  const int ln = lane & 15, lh = lane >> 4;
  const int qt = blockIdx.x, h = blockIdx.y, b = blockIdx.z;
  const int q0 = qt * 128 + w * 16;
  const float scale = 0.125f;  // 1/sqrt(64)

  const __bf16* qrow = Q + (size_t)(b * SEQ + q0 + ln) * DM + h * DH;
  v16bf qf0 = load_fragA(qrow, lh);
  v16bf qf1 = load_fragA(qrow + 32, lh);

  const __bf16* kbase = Kg + (size_t)(b * SEQ) * DM + h * DH;
  const __bf16* vtb = Vt + (size_t)(b * NH + h) * DH * SEQ;

  const int krow = tid >> 3, kch = tid & 7;   // K: 128 rows x 8 chunks / 4 rounds
  const int vrow = tid >> 4, vch = tid & 15;  // V: 64 rows x 16 chunks / 4 rounds
  auto stageKV = [&](int kt, int buf) {
    const __bf16* gk = kbase + (size_t)(kt * 128) * DM;
#pragma unroll
    for (int t = 0; t < 4; ++t)
      async_ld128(&sK[buf][(krow + t * 32) * 72 + kch * 8],
                  gk + (size_t)(krow + t * 32) * DM + kch * 8);
    const __bf16* gv = vtb + kt * 128;
#pragma unroll
    for (int t = 0; t < 4; ++t)
      async_ld128(&sV[buf][(vrow + t * 16) * 136 + vch * 8],
                  gv + (size_t)(vrow + t * 16) * SEQ + vch * 8);
  };

  const v8f vz = {0.f, 0.f, 0.f, 0.f, 0.f, 0.f, 0.f, 0.f};
  v8f o[4];
#pragma unroll
  for (int d = 0; d < 4; ++d) o[d] = vz;
  float mrow[8], lrow[8];
#pragma unroll
  for (int r = 0; r < 8; ++r) { mrow[r] = -1e30f; lrow[r] = 0.f; }

  __bf16* pw = &sP[w * 16 * 136];

  stageKV(0, 0);
  for (int kt = 0; kt < SEQ / 128; ++kt) {
    const int buf = kt & 1;
    wait_async0();
    __syncthreads();
    if (kt + 1 < SEQ / 128) stageKV(kt + 1, buf ^ 1);

    // S = Q K^T (16 q x 128 keys per wave)
    v8f s[8];
#pragma unroll
    for (int n = 0; n < 8; ++n) {
      v8f a = vz;
      v16bf k0 = load_fragB(&sK[buf][(n * 16 + ln) * 72], lh);
      a = wmma_bf16(qf0, k0, a);
      v16bf k1 = load_fragB(&sK[buf][(n * 16 + ln) * 72 + 32], lh);
      a = wmma_bf16(qf1, k1, a);
      s[n] = a;
    }

    // online softmax (row M = lh*8 + r; 16 lanes of same half share each row)
    float sf[8];
#pragma unroll
    for (int r = 0; r < 8; ++r) {
      float rm = -1e30f;
#pragma unroll
      for (int n = 0; n < 8; ++n) rm = fmaxf(rm, s[n][r] * scale);
      rm = fmaxf(rm, __shfl_xor(rm, 1, 32));
      rm = fmaxf(rm, __shfl_xor(rm, 2, 32));
      rm = fmaxf(rm, __shfl_xor(rm, 4, 32));
      rm = fmaxf(rm, __shfl_xor(rm, 8, 32));
      float mn = fmaxf(mrow[r], rm);
      sf[r] = __expf(mrow[r] - mn);
      mrow[r] = mn;
      float rs = 0.f;
#pragma unroll
      for (int n = 0; n < 8; ++n) {
        float p = __expf(s[n][r] * scale - mn);
        rs += p;
        pw[(lh * 8 + r) * 136 + n * 16 + ln] = (__bf16)p;
      }
      rs += __shfl_xor(rs, 1, 32);
      rs += __shfl_xor(rs, 2, 32);
      rs += __shfl_xor(rs, 4, 32);
      rs += __shfl_xor(rs, 8, 32);
      lrow[r] = lrow[r] * sf[r] + rs;
    }
#pragma unroll
    for (int d = 0; d < 4; ++d)
#pragma unroll
      for (int r = 0; r < 8; ++r) o[d][r] *= sf[r];
    __syncthreads();  // make P visible across lanes

    // O += P * V
#pragma unroll
    for (int kk = 0; kk < 4; ++kk) {
      v16bf pf = load_fragA(pw + ln * 136 + kk * 32, lh);
#pragma unroll
      for (int d = 0; d < 4; ++d) {
        v16bf vf = load_fragB(&sV[buf][(d * 16 + ln) * 136 + kk * 32], lh);
        o[d] = wmma_bf16(pf, vf, o[d]);
      }
    }
  }

#pragma unroll
  for (int r = 0; r < 8; ++r) {
    const float inv = 1.f / lrow[r];
#pragma unroll
    for (int d = 0; d < 4; ++d)
      O[(size_t)(b * SEQ + q0 + lh * 8 + r) * DM + h * DH + d * 16 + ln] =
          (__bf16)(o[d][r] * inv);
  }
}

// ---------- launch ----------
extern "C" void kernel_launch(void* const* d_in, const int* in_sizes, int n_in,
                              void* d_out, int out_size, void* d_ws, size_t ws_size,
                              hipStream_t stream) {
  (void)in_sizes; (void)n_in; (void)out_size; (void)ws_size;
  const float* x  = (const float*)d_in[0];
  const float* wq = (const float*)d_in[1];
  const float* bq = (const float*)d_in[2];
  const float* wk = (const float*)d_in[3];
  const float* bk = (const float*)d_in[4];
  const float* wv = (const float*)d_in[5];
  const float* bv = (const float*)d_in[6];
  const float* wo = (const float*)d_in[7];
  const float* bo = (const float*)d_in[8];

  char* ws = (char*)d_ws;
  __bf16* xb  = (__bf16*)(ws + 0);                  //  8 MiB
  __bf16* wqT = (__bf16*)(ws + 8388608);            //  2 MiB
  __bf16* wkT = (__bf16*)(ws + 10485760);
  __bf16* wvT = (__bf16*)(ws + 12582912);
  __bf16* woT = (__bf16*)(ws + 14680064);
  __bf16* Qg  = (__bf16*)(ws + 16777216);           //  8 MiB
  __bf16* Kk  = (__bf16*)(ws + 25165824);           //  8 MiB
  __bf16* Vtg = (__bf16*)(ws + 33554432);           //  8 MiB
  __bf16* Og  = (__bf16*)(ws + 41943040);           //  8 MiB (total 48 MiB)

  cvt_kernel<<<(MTOT * DM) / 256, 256, 0, stream>>>(x, xb, MTOT * DM);
  cvtT_kernel<<<(DM * DM) / 256, 256, 0, stream>>>(wq, wqT);
  cvtT_kernel<<<(DM * DM) / 256, 256, 0, stream>>>(wk, wkT);
  cvtT_kernel<<<(DM * DM) / 256, 256, 0, stream>>>(wv, wvT);
  cvtT_kernel<<<(DM * DM) / 256, 256, 0, stream>>>(wo, woT);

  dim3 gg(DM / 128, MTOT / 128);  // (8, 32)
  gemm_bf16_kernel<<<gg, 256, 0, stream>>>(xb, wqT, bq, Qg, 0);
  gemm_bf16_kernel<<<gg, 256, 0, stream>>>(xb, wkT, bk, Kk, 0);
  gemm_bf16_kernel<<<gg, 256, 0, stream>>>(xb, wvT, bv, Vtg, 2);

  dim3 ga(SEQ / 128, NH, NB);
  attn_kernel<<<ga, 256, 0, stream>>>(Qg, Kk, Vtg, Og);

  gemm_bf16_kernel<<<gg, 256, 0, stream>>>(Og, woT, bo, d_out, 1);
}
